// CapsNet_59768764891936
// MI455X (gfx1250) — matrix operand order repaired and checked by reading, not compiled, not run on hardware
//
#include <hip/hip_runtime.h>
#include <hip/hip_bf16.h>
#include <math.h>

typedef __attribute__((ext_vector_type(16))) _Float16 v16h;
typedef __attribute__((ext_vector_type(8)))  _Float16 v8h;
typedef __attribute__((ext_vector_type(8)))  float    v8f;

#define CAPS_EPS 1e-8f

// ---------------------------------------------------------------------------
// conv1: [256,1,28,28] * [256,1,9,9] stride1 VALID -> relu -> NHWC f16
// block = one (b,y,x), thread = co (256)
// ---------------------------------------------------------------------------
__global__ void conv1_kernel(const float* __restrict__ x,
                             const float* __restrict__ w1,
                             const float* __restrict__ b1,
                             _Float16* __restrict__ h1t) {
  int bid = blockIdx.x;             // b*400 + oy*20 + ox
  int b   = bid / 400;
  int p   = bid % 400;
  int oy  = p / 20, ox = p % 20;
  __shared__ float patch[81];
  int tid = threadIdx.x;
  if (tid < 81) {
    int ky = tid / 9, kx = tid % 9;
    patch[tid] = x[((size_t)b * 28 + (oy + ky)) * 28 + (ox + kx)];
  }
  __syncthreads();
  int co = tid;
  const float* w = w1 + co * 81;
  float acc = b1[co];
  #pragma unroll 9
  for (int t = 0; t < 81; t++) acc += patch[t] * w[t];
  acc = fmaxf(acc, 0.0f);
  h1t[(size_t)bid * 256 + co] = (_Float16)acc;   // [b,y,x,co] NHWC
}

// ---------------------------------------------------------------------------
// conv2 weight transpose: w2[co,ci,ky,kx] f32 -> wt2[tap][co][ci] f16
// ---------------------------------------------------------------------------
__global__ void w2t_kernel(const float* __restrict__ w2,
                           _Float16* __restrict__ wt2) {
  int i = blockIdx.x * 256 + threadIdx.x;       // 81*256*256 = 5,308,416
  if (i >= 81 * 256 * 256) return;
  int tap = i >> 16;
  int rem = i & 65535;
  int co = rem >> 8, ci = rem & 255;
  wt2[i] = (_Float16)w2[((size_t)co * 256 + ci) * 81 + tap];
}

// ---------------------------------------------------------------------------
// WMMA fragment loaders (16-bit A 16x32 / B 32x16 layouts per CDNA5 ISA)
// LDS row stride = 40 f16 (80B, 16B aligned, conflict-free)
// ---------------------------------------------------------------------------
__device__ __forceinline__ v16h frag_a(const _Float16* lds, int row, int lane) {
  union { v16h v; v8h h[2]; } u;
  int kb = (lane >> 4) << 3;                    // 0 or 8
  const _Float16* p = lds + row * 40;
  u.h[0] = *(const v8h*)(p + kb);
  u.h[1] = *(const v8h*)(p + 16 + kb);
  return u.v;
}
__device__ __forceinline__ v16h frag_b(const _Float16* lds, int col, int lane) {
  union { v16h v; v8h h[2]; } u;
  int kb = (lane >> 4) << 3;                    // 0 or 8
  const _Float16* p = lds + col * 40 + kb;
  u.h[0] = *(const v8h*)(p);
  u.h[1] = *(const v8h*)(p + 16);
  return u.v;
}

// ---------------------------------------------------------------------------
// conv2 implicit GEMM via WMMA f16 (f32 accum):
//   M=(b,oy,ox)=9216, N=co=256, K=(ky,kx,ci)=81*256
//   A[m][k]  = h1t[b, oy*2+ky, ox*2+kx, ci]   (NHWC -> contiguous over ci)
//   Bt[n][k] = wt2[tap, co, ci]               (contiguous over ci)
// tile 128x64, 256 threads (8 waves, 4x2), 2x2 WMMA tiles per wave.
// epilogue: +bias, relu, store f32 to h2[b][co*36 + oy*6 + ox]
// ---------------------------------------------------------------------------
__global__ void conv2_wmma_kernel(const _Float16* __restrict__ h1t,
                                  const _Float16* __restrict__ wt2,
                                  const float* __restrict__ b2,
                                  float* __restrict__ h2) {
  __shared__ _Float16 lA[128 * 40];
  __shared__ _Float16 lB[64 * 40];
  int tid  = threadIdx.x;
  int lane = tid & 31, wave = tid >> 5;
  int wm = wave & 3, wn = wave >> 2;
  int m0 = blockIdx.x * 128;
  int n0 = blockIdx.y * 64;

  // A-tile fill assignment: each thread copies 16 f16 of one row
  int rowA = tid >> 1, halfA = tid & 1;
  int mA = m0 + rowA;
  int bA = mA / 36, r36A = mA % 36;
  int oyA = r36A / 6, oxA = r36A % 6;
  // B-tile fill assignment: each thread copies 8 f16 of one row
  int rowB = tid >> 2, qB = tid & 3;
  int coB = n0 + rowB;

  v8f zero = {0.f,0.f,0.f,0.f,0.f,0.f,0.f,0.f};
  v8f acc[2][2];
  acc[0][0] = zero; acc[0][1] = zero; acc[1][0] = zero; acc[1][1] = zero;

  for (int kk = 0; kk < 648; kk++) {            // 81 taps * 8 ci-chunks of 32
    int tap = kk >> 3, kc = kk & 7;
    int ky = tap / 9, kx = tap % 9;
    __syncthreads();
    {
      int iy = oyA * 2 + ky, ix = oxA * 2 + kx;
      const v8h* s = (const v8h*)(h1t +
          ((((size_t)bA * 20 + iy) * 20 + ix) << 8) + kc * 32 + halfA * 16);
      v8h* d = (v8h*)(lA + rowA * 40 + halfA * 16);
      d[0] = s[0]; d[1] = s[1];
    }
    {
      const v8h* s = (const v8h*)(wt2 +
          (((size_t)tap * 256 + coB) << 8) + kc * 32 + qB * 8);
      *(v8h*)(lB + rowB * 40 + qB * 8) = *s;
    }
    __syncthreads();

    int ar = wm * 32 + (lane & 15);
    int bc = wn * 32 + (lane & 15);
    v16h a0 = frag_a(lA, ar,      lane);
    v16h a1 = frag_a(lA, ar + 16, lane);
    v16h b0 = frag_b(lB, bc,      lane);
    v16h b1 = frag_b(lB, bc + 16, lane);
    acc[0][0] = __builtin_amdgcn_wmma_f32_16x16x32_f16(false, a0, false, b0, (short)0, acc[0][0], false, false);
    acc[0][1] = __builtin_amdgcn_wmma_f32_16x16x32_f16(false, a0, false, b1, (short)0, acc[0][1], false, false);
    acc[1][0] = __builtin_amdgcn_wmma_f32_16x16x32_f16(false, a1, false, b0, (short)0, acc[1][0], false, false);
    acc[1][1] = __builtin_amdgcn_wmma_f32_16x16x32_f16(false, a1, false, b1, (short)0, acc[1][1], false, false);
  }

  int lr = lane & 15, lh = lane >> 4;
  #pragma unroll
  for (int i = 0; i < 2; i++)
    #pragma unroll
    for (int j = 0; j < 2; j++) {
      v8f a = acc[i][j];
      #pragma unroll
      for (int r = 0; r < 8; r++) {
        int m  = m0 + wm * 32 + i * 16 + r + lh * 8;
        int co = n0 + wn * 32 + j * 16 + lr;
        int b = m / 36, r36 = m % 36;
        float val = fmaxf(a[r] + b2[co], 0.0f);
        h2[(size_t)b * 9216 + co * 36 + r36] = val;
      }
    }
}

// ---------------------------------------------------------------------------
// squash over groups of 8 (in place on h2): one thread per capsule
// ---------------------------------------------------------------------------
__global__ void squash_kernel(float* __restrict__ h2, int ncaps) {
  int i = blockIdx.x * 256 + threadIdx.x;
  if (i >= ncaps) return;
  float* p = h2 + (size_t)i * 8;
  float v[8], sq = 0.f;
  #pragma unroll
  for (int d = 0; d < 8; d++) { v[d] = p[d]; sq += v[d] * v[d]; }
  float sc = (sq / (1.0f + sq)) / sqrtf(sq + CAPS_EPS);
  #pragma unroll
  for (int d = 0; d < 8; d++) p[d] = v[d] * sc;
}

// ---------------------------------------------------------------------------
// u_hat[b,c,n,f] = sum_d u[b,n,d] * W[c,n,d,f]   (K=8; VALU f32, store f16)
// ---------------------------------------------------------------------------
__global__ void uhat_kernel(const float* __restrict__ u,
                            const float* __restrict__ W,
                            _Float16* __restrict__ uh) {
  int n = blockIdx.x * 128 + threadIdx.x;       // 0..1151
  int c = blockIdx.y, b = blockIdx.z;
  float ud[8];
  const float* up = u + ((size_t)b * 1152 + n) * 8;
  #pragma unroll
  for (int d = 0; d < 8; d++) ud[d] = up[d];
  const float* wp = W + ((size_t)c * 1152 + n) * 128;
  _Float16* op = uh + ((size_t)(b * 10 + c) * 1152 + n) * 16;
  #pragma unroll
  for (int f = 0; f < 16; f++) {
    float a = 0.f;
    #pragma unroll
    for (int d = 0; d < 8; d++) a += ud[d] * wp[d * 16 + f];
    op[f] = (_Float16)a;
  }
}

__global__ void zero_kernel(float* __restrict__ p, int n) {
  int i = blockIdx.x * 256 + threadIdx.x;
  if (i < n) p[i] = 0.f;
}

// ---------------------------------------------------------------------------
// one routing iteration; workgroup per (c=blockIdx.x, b=blockIdx.y)
// ---------------------------------------------------------------------------
__global__ void routing_kernel(const _Float16* __restrict__ uh,
                               float* __restrict__ blog,
                               float* __restrict__ vbuf,
                               float* __restrict__ cls,
                               int last) {
  int c = blockIdx.x, b = blockIdx.y, tid = threadIdx.x;
  __shared__ float csh[1152];
  __shared__ float red[256];
  __shared__ float s_sh[16];
  __shared__ float v_sh[16];
  size_t bc = (size_t)b * 10 + c;
  float* bl = blog + bc * 1152;
  const _Float16* uhp = uh + bc * 1152 * 16;

  // softmax over n: max
  float mx = -1e30f;
  for (int n = tid; n < 1152; n += 256) mx = fmaxf(mx, bl[n]);
  red[tid] = mx; __syncthreads();
  for (int s = 128; s > 0; s >>= 1) {
    if (tid < s) red[tid] = fmaxf(red[tid], red[tid + s]);
    __syncthreads();
  }
  mx = red[0]; __syncthreads();
  // exp + sum
  float ls = 0.f;
  for (int n = tid; n < 1152; n += 256) {
    float e = __expf(bl[n] - mx); csh[n] = e; ls += e;
  }
  red[tid] = ls; __syncthreads();
  for (int s = 128; s > 0; s >>= 1) {
    if (tid < s) red[tid] += red[tid + s];
    __syncthreads();
  }
  float inv = 1.0f / red[0];
  if (tid < 16) s_sh[tid] = 0.f;
  __syncthreads();
  // s[f] = sum_n c_n * u_hat[n,f]
  float ps[16];
  #pragma unroll
  for (int f = 0; f < 16; f++) ps[f] = 0.f;
  for (int n = tid; n < 1152; n += 256) {
    float w = csh[n] * inv;
    const _Float16* up = uhp + (size_t)n * 16;
    #pragma unroll
    for (int f = 0; f < 16; f++) ps[f] += w * (float)up[f];
  }
  #pragma unroll
  for (int f = 0; f < 16; f++) atomicAdd(&s_sh[f], ps[f]);
  __syncthreads();
  if (tid == 0) {
    float sq = 0.f;
    #pragma unroll
    for (int f = 0; f < 16; f++) sq += s_sh[f] * s_sh[f];
    float sc = (sq / (1.0f + sq)) / sqrtf(sq + CAPS_EPS);
    float vq = 0.f;
    #pragma unroll
    for (int f = 0; f < 16; f++) {
      float vv = s_sh[f] * sc; v_sh[f] = vv; vq += vv * vv;
    }
    if (last) cls[bc] = sqrtf(vq + CAPS_EPS);
  }
  __syncthreads();
  if (last) {
    if (tid < 16) vbuf[bc * 16 + tid] = v_sh[tid];
  } else {
    for (int n = tid; n < 1152; n += 256) {
      const _Float16* up = uhp + (size_t)n * 16;
      float dot = 0.f;
      #pragma unroll
      for (int f = 0; f < 16; f++) dot += v_sh[f] * (float)up[f];
      bl[n] += dot;
    }
  }
}

// ---------------------------------------------------------------------------
// argmax mask -> decoder input f16 [256,160]; one thread per batch element
// ---------------------------------------------------------------------------
__global__ void mask_kernel(const float* __restrict__ cls,
                            const float* __restrict__ vbuf,
                            _Float16* __restrict__ rf16) {
  int b = threadIdx.x;
  const float* p = cls + (size_t)b * 10;
  int idx = 0; float best = p[0];
  for (int c = 1; c < 10; c++) if (p[c] > best) { best = p[c]; idx = c; }
  for (int i = 0; i < 160; i++) {
    int c = i >> 4, f = i & 15;
    float vv = (c == idx) ? vbuf[((size_t)b * 10 + c) * 16 + f] : 0.f;
    rf16[(size_t)b * 160 + i] = (_Float16)vv;
  }
}

// generic f32[K,N] -> f16[N,K] transpose
__global__ void transpose_f16_kernel(const float* __restrict__ src,
                                     _Float16* __restrict__ dst,
                                     int K, int N) {
  int i = blockIdx.x * 256 + threadIdx.x;
  if (i >= K * N) return;
  int n = i / K, k = i % K;
  dst[i] = (_Float16)src[(size_t)k * N + n];
}

// ---------------------------------------------------------------------------
// generic FC via WMMA f16: out = act(A[256,K] x W[K,N] + bias)
// A f16 row-major, Wt f16 [N,K]; act 0: relu->f16, 1: sigmoid->f32
// ---------------------------------------------------------------------------
__global__ void fc_wmma_kernel(const _Float16* __restrict__ A,
                               const _Float16* __restrict__ Wt,
                               const float* __restrict__ bias,
                               _Float16* __restrict__ outh,
                               float* __restrict__ outf,
                               int N, int K, int act) {
  __shared__ _Float16 lA[128 * 40];
  __shared__ _Float16 lB[64 * 40];
  int tid = threadIdx.x;
  int lane = tid & 31, wave = tid >> 5;
  int wm = wave & 3, wn = wave >> 2;
  int m0 = blockIdx.x * 128, n0 = blockIdx.y * 64;
  int rowA = tid >> 1, halfA = tid & 1;
  int rowB = tid >> 2, qB = tid & 3;
  int nB = n0 + rowB;

  v8f zero = {0.f,0.f,0.f,0.f,0.f,0.f,0.f,0.f};
  v8f acc[2][2];
  acc[0][0] = zero; acc[0][1] = zero; acc[1][0] = zero; acc[1][1] = zero;

  int nk = K >> 5;
  for (int kk = 0; kk < nk; kk++) {
    int k0 = kk << 5;
    __syncthreads();
    {
      const v8h* s = (const v8h*)(A + (size_t)(m0 + rowA) * K + k0 + halfA * 16);
      v8h* d = (v8h*)(lA + rowA * 40 + halfA * 16);
      d[0] = s[0]; d[1] = s[1];
    }
    {
      v8h val;
      if (nB < N) {
        val = *(const v8h*)(Wt + (size_t)nB * K + k0 + qB * 8);
      } else {
        #pragma unroll
        for (int z = 0; z < 8; z++) val[z] = (_Float16)0.f;
      }
      *(v8h*)(lB + rowB * 40 + qB * 8) = val;
    }
    __syncthreads();

    int ar = wm * 32 + (lane & 15);
    int bc = wn * 32 + (lane & 15);
    v16h a0 = frag_a(lA, ar,      lane);
    v16h a1 = frag_a(lA, ar + 16, lane);
    v16h b0 = frag_b(lB, bc,      lane);
    v16h b1 = frag_b(lB, bc + 16, lane);
    acc[0][0] = __builtin_amdgcn_wmma_f32_16x16x32_f16(false, a0, false, b0, (short)0, acc[0][0], false, false);
    acc[0][1] = __builtin_amdgcn_wmma_f32_16x16x32_f16(false, a0, false, b1, (short)0, acc[0][1], false, false);
    acc[1][0] = __builtin_amdgcn_wmma_f32_16x16x32_f16(false, a1, false, b0, (short)0, acc[1][0], false, false);
    acc[1][1] = __builtin_amdgcn_wmma_f32_16x16x32_f16(false, a1, false, b1, (short)0, acc[1][1], false, false);
  }

  int lr = lane & 15, lh = lane >> 4;
  #pragma unroll
  for (int i = 0; i < 2; i++)
    #pragma unroll
    for (int j = 0; j < 2; j++) {
      v8f a = acc[i][j];
      #pragma unroll
      for (int r = 0; r < 8; r++) {
        int m = m0 + wm * 32 + i * 16 + r + lh * 8;
        int n = n0 + wn * 32 + j * 16 + lr;
        if (n < N) {
          float val = a[r] + bias[n];
          if (act == 0) {
            val = fmaxf(val, 0.f);
            outh[(size_t)m * N + n] = (_Float16)val;
          } else {
            val = 1.0f / (1.0f + __expf(-val));
            outf[(size_t)m * N + n] = val;
          }
        }
      }
    }
}

// ---------------------------------------------------------------------------
extern "C" void kernel_launch(void* const* d_in, const int* in_sizes, int n_in,
                              void* d_out, int out_size, void* d_ws, size_t ws_size,
                              hipStream_t stream) {
  (void)in_sizes; (void)n_in; (void)out_size; (void)ws_size;
  const float* x   = (const float*)d_in[0];
  const float* w1  = (const float*)d_in[1];
  const float* b1  = (const float*)d_in[2];
  const float* w2  = (const float*)d_in[3];
  const float* b2  = (const float*)d_in[4];
  const float* W   = (const float*)d_in[5];
  const float* dw1 = (const float*)d_in[6];
  const float* db1 = (const float*)d_in[7];
  const float* dw2 = (const float*)d_in[8];
  const float* db2 = (const float*)d_in[9];
  const float* dw3 = (const float*)d_in[10];
  const float* db3 = (const float*)d_in[11];

  char* ws = (char*)d_ws;
  size_t off = 0;
  auto alloc = [&](size_t bytes) {
    void* p = ws + off;
    off += (bytes + 255) & ~(size_t)255;
    return p;
  };
  _Float16* h1t  = (_Float16*)alloc(26214400ull * 2);  // [256,20,20,256] f16
  _Float16* wt2  = (_Float16*)alloc(5308416ull  * 2);  // [81,256,256]    f16
  float*    h2   = (float*)   alloc(2359296ull  * 4);  // [256,9216]      f32 (u in place)
  _Float16* uh   = (_Float16*)alloc(47185920ull * 2);  // [256,10,1152,16] f16
  float*    blog = (float*)   alloc(2949120ull  * 4);  // [256,10,1152]   f32
  float*    vbuf = (float*)   alloc(40960ull    * 4);  // [256,10,16]     f32
  _Float16* rf   = (_Float16*)alloc(40960ull    * 2);  // [256,160]       f16
  _Float16* w1t  = (_Float16*)alloc(81920ull    * 2);  // [512,160]       f16
  _Float16* w2td = (_Float16*)alloc(524288ull   * 2);  // [1024,512]      f16
  _Float16* w3t  = (_Float16*)alloc(802816ull   * 2);  // [784,1024]      f16
  _Float16* a1   = (_Float16*)alloc(131072ull   * 2);  // [256,512]       f16
  _Float16* a2   = (_Float16*)alloc(262144ull   * 2);  // [256,1024]      f16

  float* cls = (float*)d_out;           // [256,10]
  float* rec = (float*)d_out + 2560;    // [256,784]

  conv1_kernel<<<102400, 256, 0, stream>>>(x, w1, b1, h1t);
  w2t_kernel<<<20736, 256, 0, stream>>>(w2, wt2);
  conv2_wmma_kernel<<<dim3(72, 4), 256, 0, stream>>>(h1t, wt2, b2, h2);
  squash_kernel<<<1152, 256, 0, stream>>>(h2, 294912);
  uhat_kernel<<<dim3(9, 10, 256), 128, 0, stream>>>(h2, W, uh);
  zero_kernel<<<11520, 256, 0, stream>>>(blog, 2949120);
  routing_kernel<<<dim3(10, 256), 256, 0, stream>>>(uh, blog, vbuf, cls, 0);
  routing_kernel<<<dim3(10, 256), 256, 0, stream>>>(uh, blog, vbuf, cls, 0);
  routing_kernel<<<dim3(10, 256), 256, 0, stream>>>(uh, blog, vbuf, cls, 1);
  mask_kernel<<<1, 256, 0, stream>>>(cls, vbuf, rf);
  transpose_f16_kernel<<<(81920 + 255) / 256, 256, 0, stream>>>(dw1, w1t, 160, 512);
  transpose_f16_kernel<<<(524288 + 255) / 256, 256, 0, stream>>>(dw2, w2td, 512, 1024);
  transpose_f16_kernel<<<(802816 + 255) / 256, 256, 0, stream>>>(dw3, w3t, 1024, 784);
  fc_wmma_kernel<<<dim3(2, 8),  256, 0, stream>>>(rf, w1t,  db1, a1, nullptr, 512, 160, 0);
  fc_wmma_kernel<<<dim3(2, 16), 256, 0, stream>>>(a1, w2td, db2, a2, nullptr, 1024, 512, 0);
  fc_wmma_kernel<<<dim3(2, 13), 256, 0, stream>>>(a2, w3t,  db3, nullptr, rec, 784, 1024, 1);
}
